// SoftSymmetricAlignmentHead_47072841564314
// MI455X (gfx1250) — compile-verified, matrix operand order checked by reading, not run on hardware
//
#include <hip/hip_runtime.h>
#include <hip/hip_bf16.h>

typedef __attribute__((ext_vector_type(16))) _Float16 v16h;
typedef __attribute__((ext_vector_type(8)))  float    v8f;
typedef __attribute__((ext_vector_type(4)))  unsigned int v4u;
typedef __attribute__((ext_vector_type(8)))  int      v8i;
typedef __attribute__((ext_vector_type(4)))  int      v4i;

#define LL   512
#define DD   64
#define NPAIR 4

#if __has_include(<hip/amd_detail/amd_gfx1250_TDM.h>)
#define SSA_TDM_SIX_ARG 1
#endif

// d-index held by (k-half h, lane-half half, vgpr v), per CDNA5 16-bit A 16x32 layout
__device__ __forceinline__ int kbase(int h, int half, int v) {
  return h * 32 + ((v < 4) ? 0 : 16) + half * 8 + (v & 3) * 2;
}

// TDM: DMA one 16x64-f32 tile (rows contiguous, row stride 64 elems) from global
// memory into LDS at byte offset lds_off. D# built per CDNA5 ISA 8.3/8.4.
__device__ __forceinline__ void tdm_load_tile(unsigned int lds_off, const float* gsrc) {
  unsigned long long ga = (unsigned long long)(uintptr_t)gsrc;
  v4u g0;
  g0.x = 1u;                                                   // count=1, user mode
  g0.y = lds_off;                                              // D#.lds_addr
  g0.z = (unsigned int)ga;                                     // global_addr[31:0]
  g0.w = (unsigned int)((ga >> 32) & 0x1FFFFFFu) | (2u << 30); // addr[56:32] | type=2
  v8i g1;
  g1[0] = (int)(2u << 16);        // data_size = 4 bytes
  g1[1] = (int)(64u << 16);       // tensor_dim0 = 64 (bits 79:48)
  g1[2] = (int)(512u << 16);      // tensor_dim0 hi = 0 | tensor_dim1 = 512
  g1[3] = (int)(64u << 16);       // tensor_dim1 hi = 0 | tile_dim0 = 64
  g1[4] = (int)16u;               // tile_dim1 = 16, tile_dim2 = 0 (2-D tile)
  g1[5] = (int)64u;               // tensor_dim0_stride = 64
  g1[6] = 0;                      // stride0 hi | stride1 lo (unused for 2-D)
  g1[7] = 0;
  v4i z4 = {0, 0, 0, 0};
#if defined(SSA_TDM_SIX_ARG)
  v8i z8 = {0, 0, 0, 0, 0, 0, 0, 0};
  __builtin_amdgcn_tensor_load_to_lds(g0, g1, z4, z4, z8, 0);
#else
  __builtin_amdgcn_tensor_load_to_lds(g0, g1, z4, z4, 0);
#endif
}

// Compute one column of the L1-distance tile via WMMA (B = ones, hi/lo f16 split).
// Returns v8f where element r = sum_d |x0 - yrow| for row M = r + 8*half.
__device__ __forceinline__ v8f l1_col_wmma(const float2 x0[2][8], const float* yr,
                                           int half, const v16h& ones) {
  v8f c = {0.f, 0.f, 0.f, 0.f, 0.f, 0.f, 0.f, 0.f};
#pragma unroll
  for (int h = 0; h < 2; ++h) {
    v16h ahi, alo;
#pragma unroll
    for (int v = 0; v < 8; ++v) {
      float2 x = x0[h][v];
      float2 y = *(const float2*)(yr + kbase(h, half, v));
      float d0 = fabsf(x.x - y.x);
      float d1 = fabsf(x.y - y.y);
      _Float16 h0 = (_Float16)d0, h1 = (_Float16)d1;
      ahi[2 * v]     = h0;
      ahi[2 * v + 1] = h1;
      alo[2 * v]     = (_Float16)(d0 - (float)h0);
      alo[2 * v + 1] = (_Float16)(d1 - (float)h1);
    }
    // D[m, *] = sum_k A[m,k] since B is all ones -> exact d-reduction in f32
    c = __builtin_amdgcn_wmma_f32_16x16x32_f16(false, ahi, false, ones, (short)0, c, false, false);
    c = __builtin_amdgcn_wmma_f32_16x16x32_f16(false, alo, false, ones, (short)0, c, false, false);
  }
  return c;
}

// Online softmax stats (max, sum-exp) over the "key" axis, per "query" row.
// grid: (32 row-tiles, 4 pairs), block: 256 (8 wave32).
__global__ __launch_bounds__(256) void ssa_stats(const float* __restrict__ X,
                                                 const float* __restrict__ Y,
                                                 float* __restrict__ omax,
                                                 float* __restrict__ osum) {
  const int p    = blockIdx.y;
  const int m0   = blockIdx.x * 16;
  const int tid  = threadIdx.x;
  const int lane = tid & 31, wave = tid >> 5;
  const int half = lane >> 4, mrow = tid & 15;
  const float* Xb = X + (size_t)p * LL * DD;
  const float* Yb = Y + (size_t)p * LL * DD;

  __shared__ __align__(128) float ytile[8][16 * DD];   // wave-private TDM landing pads
  const unsigned int yoff = (unsigned int)(uintptr_t)&ytile[wave][0];

  // Preload this lane's query-row slice (its WMMA-A d-set) into registers.
  float2 x0[2][8];
  const float* xr = Xb + (size_t)(m0 + mrow) * DD;
#pragma unroll
  for (int h = 0; h < 2; ++h)
#pragma unroll
    for (int v = 0; v < 8; ++v)
      x0[h][v] = *(const float2*)(xr + kbase(h, half, v));

  v16h ones;
#pragma unroll
  for (int i = 0; i < 16; ++i) ones[i] = (_Float16)1.0f;

  float rm[8], rs[8];
#pragma unroll
  for (int r = 0; r < 8; ++r) { rm[r] = -__builtin_inff(); rs[r] = 0.f; }

  for (int t = wave; t < 32; t += 8) {      // key tiles for this wave
    tdm_load_tile(yoff, Yb + (size_t)t * 16 * DD);     // tensor_load_to_lds
    __builtin_amdgcn_s_wait_tensorcnt(0);              // s_wait_tensorcnt 0
#pragma unroll 1
    for (int nn = 0; nn < 16; ++nn) {
      const float* yr = &ytile[wave][nn * DD];
      v8f c = l1_col_wmma(x0, yr, half, ones);
#pragma unroll
      for (int r = 0; r < 8; ++r) {
        float sv = -c[r];                   // s[m, n]
        float nm = fmaxf(rm[r], sv);
        rs[r] = rs[r] * __expf(rm[r] - nm) + __expf(sv - nm);
        rm[r] = nm;
      }
    }
  }

  __shared__ float lmax[8][16];
  __shared__ float lsum[8][16];
  if ((lane & 15) == 0) {
#pragma unroll
    for (int r = 0; r < 8; ++r) {
      lmax[wave][half * 8 + r] = rm[r];
      lsum[wave][half * 8 + r] = rs[r];
    }
  }
  __syncthreads();
  if (tid < 16) {                           // deterministic cross-wave merge
    float M = -__builtin_inff();
    for (int w = 0; w < 8; ++w) M = fmaxf(M, lmax[w][tid]);
    float S = 0.f;
    for (int w = 0; w < 8; ++w) S += lsum[w][tid] * __expf(lmax[w][tid] - M);
    omax[(size_t)p * LL + m0 + tid] = M;
    osum[(size_t)p * LL + m0 + tid] = S;
  }
}

// Recompute s, form c = a + b - a*b, accumulate sum(c*s) and sum(c) per (pair, row-tile).
__global__ __launch_bounds__(256) void ssa_acc(const float* __restrict__ X,
                                               const float* __restrict__ Y,
                                               const float* __restrict__ rmax,
                                               const float* __restrict__ rsum,
                                               const float* __restrict__ cmax,
                                               const float* __restrict__ csum,
                                               float* __restrict__ pcs,
                                               float* __restrict__ pc) {
  const int p    = blockIdx.y;
  const int m0   = blockIdx.x * 16;
  const int tid  = threadIdx.x;
  const int lane = tid & 31, wave = tid >> 5;
  const int half = lane >> 4, mrow = lane & 15;
  const float* Xb = X + (size_t)p * LL * DD;
  const float* Yb = Y + (size_t)p * LL * DD;

  __shared__ __align__(128) float ytile[8][16 * DD];
  const unsigned int yoff = (unsigned int)(uintptr_t)&ytile[wave][0];

  float2 x0[2][8];
  const float* xr = Xb + (size_t)(m0 + mrow) * DD;
#pragma unroll
  for (int h = 0; h < 2; ++h)
#pragma unroll
    for (int v = 0; v < 8; ++v)
      x0[h][v] = *(const float2*)(xr + kbase(h, half, v));

  v16h ones;
#pragma unroll
  for (int i = 0; i < 16; ++i) ones[i] = (_Float16)1.0f;

  float rm[8], rs[8];
#pragma unroll
  for (int r = 0; r < 8; ++r) {
    rm[r] = rmax[(size_t)p * LL + m0 + half * 8 + r];
    rs[r] = rsum[(size_t)p * LL + m0 + half * 8 + r];
  }

  float acs = 0.f, ac = 0.f;
  for (int t = wave; t < 32; t += 8) {
    tdm_load_tile(yoff, Yb + (size_t)t * 16 * DD);
    __builtin_amdgcn_s_wait_tensorcnt(0);
#pragma unroll 1
    for (int nn = 0; nn < 16; ++nn) {
      const int n = t * 16 + nn;
      const float* yr = &ytile[wave][nn * DD];
      v8f c = l1_col_wmma(x0, yr, half, ones);
      float cm = cmax[(size_t)p * LL + n];
      float cd = csum[(size_t)p * LL + n];
#pragma unroll
      for (int r = 0; r < 8; ++r) {
        float sv = -c[r];
        float a  = __expf(sv - rm[r]) / rs[r];
        float b  = __expf(sv - cm) / cd;
        float cc = a + b - a * b;
        acs += cc * sv;
        ac  += cc;
      }
    }
  }

  // Deterministic reduction: shuffle tree within wave, ordered sum across waves.
  for (int off = 16; off; off >>= 1) {
    acs += __shfl_xor(acs, off);
    ac  += __shfl_xor(ac, off);
  }
  __shared__ float wcs[8];
  __shared__ float wc[8];
  if (lane == 0) { wcs[wave] = acs; wc[wave] = ac; }
  __syncthreads();
  if (tid == 0) {
    float tcs = 0.f, tc = 0.f;
    for (int w = 0; w < 8; ++w) { tcs += wcs[w]; tc += wc[w]; }
    // every (m,n) was accumulated by 16 redundant lanes -> exact /16
    pcs[p * 32 + blockIdx.x] = tcs * 0.0625f;
    pc [p * 32 + blockIdx.x] = tc  * 0.0625f;
  }
}

__global__ void ssa_out(const float* __restrict__ pcs, const float* __restrict__ pc,
                        const float* __restrict__ w, const float* __restrict__ bias,
                        float* __restrict__ out) {
  int t = threadIdx.x;
  if (t >= 16) return;
  int p = t >> 2, cls = t & 3;
  float cs = 0.f, cc = 0.f;
  for (int j = 0; j < 32; ++j) { cs += pcs[p * 32 + j]; cc += pc[p * 32 + j]; }
  float cval = cs / cc;
  out[t] = cval * w[cls] + bias[cls];
}

extern "C" void kernel_launch(void* const* d_in, const int* in_sizes, int n_in,
                              void* d_out, int out_size, void* d_ws, size_t ws_size,
                              hipStream_t stream) {
  const float* seqs = (const float*)d_in[0];   // (8, 512, 64) f32
  const float* w    = (const float*)d_in[1];   // (1, 4) f32
  const float* bias = (const float*)d_in[2];   // (4,) f32
  float* out = (float*)d_out;                  // 16 f32
  float* ws  = (float*)d_ws;

  // workspace layout (floats): 4*512 each + 4*32 partials -> ~33 KB total
  float* rmax = ws;
  float* rsum = ws + 2048;
  float* cmax = ws + 4096;
  float* csum = ws + 6144;
  float* pcs  = ws + 8192;
  float* pc   = ws + 8320;

  const float* z0 = seqs;
  const float* z1 = seqs + (size_t)NPAIR * LL * DD;

  dim3 grid(32, NPAIR), blk(256);
  // row stats: softmax over n for each m
  ssa_stats<<<grid, blk, 0, stream>>>(z0, z1, rmax, rsum);
  // col stats: |.| is symmetric, so swap roles -> softmax over m for each n
  ssa_stats<<<grid, blk, 0, stream>>>(z1, z0, cmax, csum);
  // fused c-weighted accumulation (deterministic partials, no atomics)
  ssa_acc<<<grid, blk, 0, stream>>>(z0, z1, rmax, rsum, cmax, csum, pcs, pc);
  ssa_out<<<1, 32, 0, stream>>>(pcs, pc, w, bias, out);
}
